// TensorParallelMultiheadAttention_26843545600570
// MI455X (gfx1250) — matrix operand order repaired and checked by reading, not compiled, not run on hardware
//
#include <hip/hip_runtime.h>
#include <hip/hip_bf16.h>

// MI455X / gfx1250, wave32. Matrix ops via v_wmma_f32_16x16x32_bf16;
// attention K/V tile staging via the Tensor Data Mover (tensor_load_to_lds,
// TENSORcnt), double-buffered so DMA overlaps WMMA compute.
//
// Fragment layouts (wave32, CDNA5 ISA 7.12.2):
//  A (16x32 bf16): lane L -> row m=L&15; elems 0..7 = K [h*8..h*8+7],
//                  elems 8..15 = K [16+h*8..], h=L>>4.
//  B (32x16 bf16): lane L -> col n=L&15; same K order as A. A "Bt" tile
//                  stored row-major [N][K] gives two 8-bf16 runs per lane.
//  C/D (16x16 f32): lane L -> col n=L&15; vgpr r -> row (r + (L>>4)*8).

typedef __bf16 bhalf;
typedef __attribute__((ext_vector_type(16))) __bf16 bf16x16;
typedef __attribute__((ext_vector_type(8)))  __bf16 bf16x8;
typedef __attribute__((ext_vector_type(4)))  __bf16 bf16x4;
typedef __attribute__((ext_vector_type(8)))  float  f32x8;
typedef __attribute__((ext_vector_type(4)))  unsigned int u32x4;
typedef __attribute__((ext_vector_type(8)))  int i32x8;
typedef __attribute__((ext_vector_type(4)))  int i32x4;

#define DEVINL __device__ __forceinline__

static constexpr int Bb = 2, SQ = 2048, Hh = 2048, NHd = 16, HN = 128;

DEVINL f32x8 wmma_bf16(bf16x16 a, bf16x16 b, f32x8 c) {
  return __builtin_amdgcn_wmma_f32_16x16x32_bf16(false, a, false, b, (short)0, c, false, false);
}

DEVINL bf16x16 cat8(bf16x8 lo, bf16x8 hi) {
  return __builtin_shufflevector(lo, hi, 0,1,2,3,4,5,6,7,8,9,10,11,12,13,14,15);
}

DEVINL bf16x16 load_frag_a(const bhalf* base, int ld, int lane) {
  const int m = lane & 15, h = lane >> 4;
  const bhalf* p = base + m * ld + h * 8;
  return cat8(*(const bf16x8*)p, *(const bf16x8*)(p + 16));
}

DEVINL bf16x16 load_frag_bt(const bhalf* base, int ld, int lane) {
  const int n = lane & 15, h = lane >> 4;
  const bhalf* p = base + n * ld + h * 8;
  return cat8(*(const bf16x8*)p, *(const bf16x8*)(p + 16));
}

DEVINL int rfl(unsigned v) { return __builtin_amdgcn_readfirstlane((int)v); }

// LDS byte offset of a __shared__ object: in the LDS aperture the low 32 bits
// of the generic address are the wave-relative LDS offset.
DEVINL unsigned lds_off(const void* p) {
  return (unsigned)(unsigned long long)(uintptr_t)p;
}

// Issue a TDM load of a 2D bf16 tile: tile_d1 rows x tile_d0 elements,
// global row stride = stride_elems, rows packed contiguously into LDS.
// D# group0/group1 packed per CDNA5 ISA 8.3/8.4 (data_size=1 -> 2 bytes,
// type=2 "image", count=1, no gather/iterate/pad/multicast).
DEVINL void tdm_load_2d_bf16(const void* gptr, unsigned lds_byte_off,
                             unsigned tile_d0, unsigned tile_d1,
                             unsigned stride_elems) {
  unsigned long long ga = (unsigned long long)(uintptr_t)gptr;
  u32x4 g0;
  g0[0] = (unsigned)rfl(1u);                                   // count=1
  g0[1] = (unsigned)rfl(lds_byte_off);                         // lds_addr
  g0[2] = (unsigned)rfl((unsigned)ga);                         // global_addr lo
  g0[3] = (unsigned)rfl(((unsigned)(ga >> 32) & 0x1ffffffu) | (2u << 30));
  i32x8 g1;
  g1[0] = rfl(1u << 16);                                       // data_size=2B
  g1[1] = rfl((tile_d0 & 0xffffu) << 16);                      // tensor_dim0
  g1[2] = rfl(((tile_d0 >> 16) & 0xffffu) | ((tile_d1 & 0xffffu) << 16));
  g1[3] = rfl(((tile_d1 >> 16) & 0xffffu) | ((tile_d0 & 0xffffu) << 16)); // tile_dim0
  g1[4] = rfl(tile_d1 & 0xffffu);                              // tile_dim1 (dim2=0)
  g1[5] = rfl(stride_elems);                                   // dim0 stride lo
  g1[6] = rfl(0u);
  g1[7] = rfl(0u);
  i32x4 gz = {0, 0, 0, 0};
#if defined(__clang_major__) && (__clang_major__ >= 23)
  i32x8 gz8 = {0, 0, 0, 0, 0, 0, 0, 0};
  __builtin_amdgcn_tensor_load_to_lds(g0, g1, gz, gz, gz8, 0);
#else
  __builtin_amdgcn_tensor_load_to_lds(g0, g1, gz, gz, 0);
#endif
}

// ---------------------------------------------------------------------------
// Kernel 1: mixed = X @ Wqkv^T (X[m][k] = query[b][s][k], m = s*B + b),
// scattered into Q (scaled 1/sqrt(HN)), K (row major), V^T. 128x128x32 tiles,
// 8 waves of 32x64. Each 128-wide N block is one (head, q/k/v) segment.
// ---------------------------------------------------------------------------
__global__ __launch_bounds__(256)
void qkv_gemm_kernel(const float* __restrict__ query, const float* __restrict__ Wqkv,
                     bhalf* __restrict__ Qb, bhalf* __restrict__ Kb,
                     bhalf* __restrict__ VTb)
{
  __shared__ bhalf At[128 * 32] __attribute__((aligned(32)));
  __shared__ bhalf Bt[128 * 32] __attribute__((aligned(32)));

  const int tid = threadIdx.x, lane = tid & 31, w = tid >> 5;
  const int wm = w >> 1, wn = w & 1;
  const int m0 = blockIdx.x * 128;
  const int n0 = blockIdx.y * 128;

  const f32x8 zero8 = {0.f,0.f,0.f,0.f,0.f,0.f,0.f,0.f};
  f32x8 acc[2][4];
  #pragma unroll
  for (int mi = 0; mi < 2; ++mi)
    #pragma unroll
    for (int ni = 0; ni < 4; ++ni) acc[mi][ni] = zero8;

  for (int kk = 0; kk < Hh; kk += 32) {
    __syncthreads();
    #pragma unroll
    for (int i = 0; i < 4; ++i) {
      int slot = tid * 4 + i;
      int row = slot >> 3, c = slot & 7;
      int mg = m0 + row, bb = mg & 1, ss = mg >> 1;
      const float* ap = query + ((size_t)bb * SQ + ss) * Hh + kk + c * 4;
      const float* bp = Wqkv + (size_t)(n0 + row) * Hh + kk + c * 4;
      float4 v = *(const float4*)ap;
      bf16x4 oa = {(bhalf)v.x, (bhalf)v.y, (bhalf)v.z, (bhalf)v.w};
      *(bf16x4*)(At + row * 32 + c * 4) = oa;
      float4 u = *(const float4*)bp;
      bf16x4 ob = {(bhalf)u.x, (bhalf)u.y, (bhalf)u.z, (bhalf)u.w};
      *(bf16x4*)(Bt + row * 32 + c * 4) = ob;
      if (i == 0 && kk + 32 < Hh) {              // next-tile prefetch
        __builtin_prefetch(ap + 32, 0, 1);
        __builtin_prefetch(bp + 32, 0, 1);
      }
    }
    __syncthreads();

    bf16x16 af[2], bfr[4];
    #pragma unroll
    for (int mi = 0; mi < 2; ++mi)
      af[mi] = load_frag_a(At + (wm * 32 + mi * 16) * 32, 32, lane);
    #pragma unroll
    for (int ni = 0; ni < 4; ++ni)
      bfr[ni] = load_frag_bt(Bt + (wn * 64 + ni * 16) * 32, 32, lane);
    #pragma unroll
    for (int mi = 0; mi < 2; ++mi)
      #pragma unroll
      for (int ni = 0; ni < 4; ++ni)
        acc[mi][ni] = wmma_bf16(af[mi], bfr[ni], acc[mi][ni]);
  }

  const int head  = n0 / (3 * HN);
  const int which = (n0 - head * 3 * HN) >> 7;   // 0=q 1=k 2=v
  const float norm = 0.08838834764831845f;       // 1/sqrt(128) folded into Q
  const int nlo = lane & 15, half = lane >> 4;
  #pragma unroll
  for (int mi = 0; mi < 2; ++mi)
    #pragma unroll
    for (int ni = 0; ni < 4; ++ni)
      #pragma unroll
      for (int r = 0; r < 8; ++r) {
        int mg = m0 + wm * 32 + mi * 16 + r + half * 8;
        int d  = wn * 64 + ni * 16 + nlo;
        int bb = mg & 1, ss = mg >> 1;
        int g  = bb * NHd + head;                // group = b*NH + head
        float v = acc[mi][ni][r];
        if (which == 0)      Qb[((size_t)g * SQ + ss) * HN + d] = (bhalf)(v * norm);
        else if (which == 1) Kb[((size_t)g * SQ + ss) * HN + d] = (bhalf)v;
        else                 VTb[((size_t)g * HN + d) * SQ + ss] = (bhalf)v;
      }
}

// ---------------------------------------------------------------------------
// Kernel 2: flash attention per (group, 128 q rows). 8 waves x 16 q rows.
// K/V^T tiles (64 keys) DMA'd by the TDM into double-buffered LDS; wave 0
// issues, s_wait_tensorcnt + barrier publishes; DMA of tile i+1 overlaps the
// S/softmax/PV WMMA work on tile i. P converted C->A layout via per-wave LDS.
// ---------------------------------------------------------------------------
__global__ __launch_bounds__(256)
void attn_kernel(const bhalf* __restrict__ Qb, const bhalf* __restrict__ Kb,
                 const bhalf* __restrict__ VTb, bhalf* __restrict__ CTX)
{
  __shared__ bhalf klds[2][64 * 128] __attribute__((aligned(32)));  // [key][d]
  __shared__ bhalf vlds[2][128 * 64] __attribute__((aligned(32)));  // [d][key]
  __shared__ bhalf plds[8 * 16 * 64] __attribute__((aligned(32)));  // wave-local

  const int tid = threadIdx.x, lane = tid & 31, w = tid >> 5;
  const int g  = blockIdx.x;         // 0..31
  const int qt = blockIdx.y;         // 0..15
  const int nlo = lane & 15, half = lane >> 4;

  const bhalf* Qg = Qb  + (size_t)g * SQ * HN;
  const bhalf* Kg = Kb  + (size_t)g * SQ * HN;
  const bhalf* Vg = VTb + (size_t)g * HN * SQ;
  const int qrow = qt * 128 + w * 16;

  // Q fragments for this wave's 16 rows (loaded once, stay in VGPRs).
  bf16x16 qf[4];
  #pragma unroll
  for (int dc = 0; dc < 4; ++dc)
    qf[dc] = load_frag_a(Qg + (size_t)qrow * HN + dc * 32, HN, lane);

  const f32x8 zero8 = {0.f,0.f,0.f,0.f,0.f,0.f,0.f,0.f};
  f32x8 cacc[8];
  float mstate[8], lstate[8];
  #pragma unroll
  for (int i = 0; i < 8; ++i) { cacc[i] = zero8; mstate[i] = -1e30f; lstate[i] = 0.f; }

  bhalf* pw = plds + w * (16 * 64);
  const int NIT = SQ / 64;   // 32

  // Prologue: DMA tile 0 into buffer 0.
  if (w == 0) {
    tdm_load_2d_bf16(Kg, lds_off(&klds[0][0]), /*d0*/HN, /*d1*/64, /*stride*/HN);
    tdm_load_2d_bf16(Vg, lds_off(&vlds[0][0]), /*d0*/64, /*d1*/HN, /*stride*/SQ);
  }

  for (int it = 0; it < NIT; ++it) {
    if (w == 0) __builtin_amdgcn_s_wait_tensorcnt(0);
    __syncthreads();   // tile `it` resident; prior readers of buf (it+1)&1 done

    if (w == 0 && it + 1 < NIT) {   // overlap DMA of tile it+1 with compute
      int nb = (it + 1) & 1, kb = (it + 1) * 64;
      tdm_load_2d_bf16(Kg + (size_t)kb * HN, lds_off(&klds[nb][0]), HN, 64, HN);
      tdm_load_2d_bf16(Vg + kb,              lds_off(&vlds[nb][0]), 64, HN, SQ);
    }

    const bhalf* kl = klds[it & 1];
    const bhalf* vl = vlds[it & 1];

    // S = Q K^T (pre-scaled): 4 key tiles x 4 d chunks
    f32x8 sacc[4];
    #pragma unroll
    for (int kt = 0; kt < 4; ++kt) {
      sacc[kt] = zero8;
      #pragma unroll
      for (int dc = 0; dc < 4; ++dc) {
        bf16x16 bf = load_frag_bt(kl + (kt * 16) * 128 + dc * 32, 128, lane);
        sacc[kt] = wmma_bf16(qf[dc], bf, sacc[kt]);
      }
    }

    // Online softmax; rows of a C tile span the 16 lanes of a half-wave.
    float alpha[8];
    #pragma unroll
    for (int r = 0; r < 8; ++r) {
      float cm = fmaxf(fmaxf(sacc[0][r], sacc[1][r]), fmaxf(sacc[2][r], sacc[3][r]));
      #pragma unroll
      for (int off = 1; off < 16; off <<= 1) cm = fmaxf(cm, __shfl_xor(cm, off, 32));
      float mn = fmaxf(mstate[r], cm);
      alpha[r] = __expf(mstate[r] - mn);
      float rs = 0.f;
      #pragma unroll
      for (int kt = 0; kt < 4; ++kt) {
        float p = __expf(sacc[kt][r] - mn);
        pw[(r + half * 8) * 64 + kt * 16 + nlo] = (bhalf)p;  // C->A via LDS
        rs += p;
      }
      #pragma unroll
      for (int off = 1; off < 16; off <<= 1) rs += __shfl_xor(rs, off, 32);
      lstate[r] = lstate[r] * alpha[r] + rs;
      mstate[r] = mn;
    }
    #pragma unroll
    for (int dt = 0; dt < 8; ++dt)
      #pragma unroll
      for (int r = 0; r < 8; ++r) cacc[dt][r] *= alpha[r];

    // Wave-local LDS RAW on P: split DS counter wait, no block barrier needed.
    asm volatile("s_wait_dscnt 0" ::: "memory");

    // ctx += P @ V : 2 key chunks of 32 x 8 d tiles
    #pragma unroll
    for (int kkc = 0; kkc < 2; ++kkc) {
      bf16x16 af = load_frag_a(pw + kkc * 32, 64, lane);
      #pragma unroll
      for (int dt = 0; dt < 8; ++dt) {
        bf16x16 vf = load_frag_bt(vl + (dt * 16) * 64 + kkc * 32, 64, lane);
        cacc[dt] = wmma_bf16(af, vf, cacc[dt]);
      }
    }
  }

  // Reference quirk: ctx.reshape(nh, b, ...) reinterprets g as (i0=g>>1, i1=g&1).
  const int i0 = g >> 1, i1 = g & 1;
  #pragma unroll
  for (int r = 0; r < 8; ++r) {
    float inv = 1.0f / lstate[r];
    int s = qrow + r + half * 8;
    size_t rowoff = (size_t)(s * Bb + i1) * Hh;
    #pragma unroll
    for (int dt = 0; dt < 8; ++dt) {
      int col = i0 * HN + dt * 16 + nlo;
      CTX[rowoff + col] = (bhalf)(cacc[dt][r] * inv);
    }
  }
}

// ---------------------------------------------------------------------------
// Kernel 3: out = CTX @ Wproj^T + bproj, (sq,b)->(b,sq) folded into the store.
// ---------------------------------------------------------------------------
__global__ __launch_bounds__(256)
void proj_gemm_kernel(const bhalf* __restrict__ CTX, const float* __restrict__ Wproj,
                      const float* __restrict__ bproj, float* __restrict__ out)
{
  __shared__ bhalf At[128 * 32] __attribute__((aligned(32)));
  __shared__ bhalf Bt[128 * 32] __attribute__((aligned(32)));

  const int tid = threadIdx.x, lane = tid & 31, w = tid >> 5;
  const int wm = w >> 1, wn = w & 1;
  const int m0 = blockIdx.x * 128;
  const int n0 = blockIdx.y * 128;

  const f32x8 zero8 = {0.f,0.f,0.f,0.f,0.f,0.f,0.f,0.f};
  f32x8 acc[2][4];
  #pragma unroll
  for (int mi = 0; mi < 2; ++mi)
    #pragma unroll
    for (int ni = 0; ni < 4; ++ni) acc[mi][ni] = zero8;

  for (int kk = 0; kk < Hh; kk += 32) {
    __syncthreads();
    #pragma unroll
    for (int i = 0; i < 2; ++i) {   // A already bf16
      int slot = tid * 2 + i;
      int row = slot >> 2, c = slot & 3;
      const bhalf* ap = CTX + (size_t)(m0 + row) * Hh + kk + c * 8;
      *(uint4*)(At + row * 32 + c * 8) = *(const uint4*)ap;
      if (i == 0 && kk + 32 < Hh) __builtin_prefetch(ap + 32, 0, 1);
    }
    #pragma unroll
    for (int i = 0; i < 4; ++i) {   // B fp32 -> bf16
      int slot = tid * 4 + i;
      int row = slot >> 3, c = slot & 7;
      const float* bp = Wproj + (size_t)(n0 + row) * Hh + kk + c * 4;
      float4 u = *(const float4*)bp;
      bf16x4 ob = {(bhalf)u.x, (bhalf)u.y, (bhalf)u.z, (bhalf)u.w};
      *(bf16x4*)(Bt + row * 32 + c * 4) = ob;
      if (i == 0 && kk + 32 < Hh) __builtin_prefetch(bp + 32, 0, 1);
    }
    __syncthreads();

    bf16x16 af[2], bfr[4];
    #pragma unroll
    for (int mi = 0; mi < 2; ++mi)
      af[mi] = load_frag_a(At + (wm * 32 + mi * 16) * 32, 32, lane);
    #pragma unroll
    for (int ni = 0; ni < 4; ++ni)
      bfr[ni] = load_frag_bt(Bt + (wn * 64 + ni * 16) * 32, 32, lane);
    #pragma unroll
    for (int mi = 0; mi < 2; ++mi)
      #pragma unroll
      for (int ni = 0; ni < 4; ++ni)
        acc[mi][ni] = wmma_bf16(af[mi], bfr[ni], acc[mi][ni]);
  }

  const int nlo = lane & 15, half = lane >> 4;
  #pragma unroll
  for (int ni = 0; ni < 4; ++ni) {
    int n = n0 + wn * 64 + ni * 16 + nlo;
    float bias = bproj[n];
    #pragma unroll
    for (int mi = 0; mi < 2; ++mi)
      #pragma unroll
      for (int r = 0; r < 8; ++r) {
        int mg = m0 + wm * 32 + mi * 16 + r + half * 8;
        int bb = mg & 1, ss = mg >> 1;
        out[((size_t)bb * SQ + ss) * Hh + n] = acc[mi][ni][r] + bias;
      }
  }
}

// ---------------------------------------------------------------------------
extern "C" void kernel_launch(void* const* d_in, const int* in_sizes, int n_in,
                              void* d_out, int out_size, void* d_ws, size_t ws_size,
                              hipStream_t stream) {
  (void)in_sizes; (void)n_in; (void)out_size; (void)ws_size;
  const float* query = (const float*)d_in[0];
  // d_in[1] (key) and d_in[2] (value) are unused by the reference.
  const float* Wqkv  = (const float*)d_in[3];
  const float* Wproj = (const float*)d_in[4];
  const float* bproj = (const float*)d_in[5];
  float* out = (float*)d_out;

  // Workspace: Q | K | V^T | CTX, each 32*2048*128 bf16 = 16 MB (64 MB total).
  char* ws = (char*)d_ws;
  bhalf* Qb  = (bhalf*)(ws);
  bhalf* Kb  = (bhalf*)(ws + (size_t)16 * 1024 * 1024);
  bhalf* VTb = (bhalf*)(ws + (size_t)32 * 1024 * 1024);
  bhalf* CTX = (bhalf*)(ws + (size_t)48 * 1024 * 1024);

  dim3 g1(32, 48); qkv_gemm_kernel <<<g1, 256, 0, stream>>>(query, Wqkv, Qb, Kb, VTb);
  dim3 g2(32, 16); attn_kernel     <<<g2, 256, 0, stream>>>(Qb, Kb, VTb, CTX);
  dim3 g3(32, 16); proj_gemm_kernel<<<g3, 256, 0, stream>>>(CTX, Wproj, bproj, out);
}
